// MultiScaleRetention_15719580303688
// MI455X (gfx1250) — compile-verified, matrix operand order checked by reading, not run on hardware
//
#include <hip/hip_runtime.h>
#include <hip/hip_bf16.h>

typedef _Float16 half_t;
typedef __attribute__((ext_vector_type(16))) _Float16 v16h;
typedef __attribute__((ext_vector_type(8)))  float    v8f;
typedef __attribute__((ext_vector_type(4)))  unsigned int v4u;
typedef __attribute__((ext_vector_type(4)))  int v4i;

// Problem constants (match reference)
constexpr int Bb = 4, Ll = 4096, Dd = 1024, Hh = 16, DhC = 64;
constexpr int Mrows = Bb * Ll;        // 16384
constexpr int CL = 256;               // scan chunk length
constexpr int NC = Ll / CL;           // 16 chunks

// GEMM tiling: block tile 128(M) x 256(N), K-step 32, 8 waves of 64x64
#define TM 128
#define TN 256
#define TK 32
#define LDSK (TK + 8)   // padded LDS stride in halfs (80B rows, b128-aligned)

union FragH { v16h h; v4u u[2]; };

// ---------------- CDNA5 async global->LDS copy (16B per lane) ----------------
#if __has_builtin(__builtin_amdgcn_global_load_async_to_lds_b128)
#define HAVE_ASYNC_LDS 1
typedef __attribute__((address_space(1))) v4i* gptr_v4i;
typedef __attribute__((address_space(3))) v4i* lptr_v4i;
__device__ __forceinline__ void copy16_async(const half_t* g, half_t* l) {
  __builtin_amdgcn_global_load_async_to_lds_b128(
      (gptr_v4i)(void*)g, (lptr_v4i)(void*)l, 0, 0);
}
#else
#define HAVE_ASYNC_LDS 0
__device__ __forceinline__ void copy16_async(const half_t* g, half_t* l) {
  *(v4u*)l = *(const v4u*)g;   // sync fallback: global->VGPR->LDS
}
#endif

template <int N>
__device__ __forceinline__ void wait_async() {
#if HAVE_ASYNC_LDS
#if __has_builtin(__builtin_amdgcn_s_wait_asynccnt)
  __builtin_amdgcn_s_wait_asynccnt(N);
#else
  asm volatile("s_wait_asynccnt %0" :: "n"(N) : "memory");
#endif
#endif
}

// ---------------- fp32 -> fp16 conversion ----------------
__global__ __launch_bounds__(256)
void cvt_f32_f16_kernel(const float* __restrict__ in, half_t* __restrict__ out, int n) {
  int i = blockIdx.x * blockDim.x + threadIdx.x;
  if (i < n) out[i] = (half_t)in[i];
}

// ---------------- WMMA GEMM: C[M,N] = A[M,K] * Bw[N,K]^T ----------------
// mode 0: plain fp32 store; mode 1: SiLU epilogue (for the gate GEMM)
__global__ __launch_bounds__(256)
void gemm_wmma_f16(const half_t* __restrict__ A, const half_t* __restrict__ Bw,
                   float* __restrict__ C, int M, int N, int K, int mode) {
  __shared__ half_t As[2][TM * LDSK];   // 2 x 10240 B
  __shared__ half_t Bs[2][TN * LDSK];   // 2 x 20480 B   (total 60 KB)

  const int tid  = threadIdx.x;
  const int lane = tid & 31;
  const int wid  = tid >> 5;
  const int bm = blockIdx.x * TM;
  const int bn = blockIdx.y * TN;

  // cooperative tile-load mapping: 256 threads x 16B, 64 rows per pass
  const int lrow = tid >> 2;          // 0..63
  const int lcol = (tid & 3) << 3;    // 0,8,16,24

  // wave tile 64x64: 2 waves along M, 4 along N
  const int wm = (wid & 1) * 64;
  const int wn = (wid >> 1) * 64;
  const int frow = lane & 15;
  const int klo  = (lane < 16) ? 0 : 8;

  v8f acc[4][4];
#pragma unroll
  for (int i = 0; i < 4; ++i)
#pragma unroll
    for (int j = 0; j < 4; ++j)
#pragma unroll
      for (int e = 0; e < 8; ++e) acc[i][j][e] = 0.0f;

  const half_t* Abase = A  + (size_t)(bm + lrow) * K + lcol;
  const half_t* Bbase = Bw + (size_t)(bn + lrow) * K + lcol;

  auto loadTile = [&](int k0, int buf) {
#pragma unroll
    for (int r = 0; r < TM; r += 64)
      copy16_async(Abase + (size_t)r * K + k0, &As[buf][(lrow + r) * LDSK + lcol]);
#pragma unroll
    for (int r = 0; r < TN; r += 64)
      copy16_async(Bbase + (size_t)r * K + k0, &Bs[buf][(lrow + r) * LDSK + lcol]);
  };

  auto computeTile = [&](int buf) {
    // A fragment: lane l<16 holds row M=l, K = {klo..klo+7, klo+16..klo+23}
    FragH a[4], b[4];
#pragma unroll
    for (int i = 0; i < 4; ++i) {
      const half_t* p = &As[buf][(wm + i * 16 + frow) * LDSK + klo];
      a[i].u[0] = *(const v4u*)p;
      a[i].u[1] = *(const v4u*)(p + 16);
    }
#pragma unroll
    for (int j = 0; j < 4; ++j) {
      const half_t* p = &Bs[buf][(wn + j * 16 + frow) * LDSK + klo];
      b[j].u[0] = *(const v4u*)p;
      b[j].u[1] = *(const v4u*)(p + 16);
    }
#pragma unroll
    for (int i = 0; i < 4; ++i)
#pragma unroll
      for (int j = 0; j < 4; ++j)
        acc[i][j] = __builtin_amdgcn_wmma_f32_16x16x32_f16(
            false, a[i].h, false, b[j].h, (short)0, acc[i][j], false, false);
  };

  const int NT = K / TK;
  loadTile(0, 0);                       // prologue: tile 0 -> buf 0

  // steady state: unconditional issue-next / wait-current / compute
  for (int t = 0; t < NT - 1; ++t) {
    const int buf = t & 1;
    loadTile((t + 1) * TK, buf ^ 1);    // overlap next tile with this tile's math
    wait_async<6>();                    // tile t's 6 copies done; tile t+1 in flight
    __syncthreads();
    computeTile(buf);
    __syncthreads();                    // done reading buf before iter t+1 overwrites buf^1... (2-deep)
  }
  // peeled last tile
  wait_async<0>();
  __syncthreads();
  computeTile((NT - 1) & 1);

  // Epilogue: C/D layout — VGPR e is row e (lanes 0-15) / row e+8 (lanes 16-31), col = lane%16
#pragma unroll
  for (int i = 0; i < 4; ++i) {
    const int row0 = bm + wm + i * 16 + ((lane < 16) ? 0 : 8);
#pragma unroll
    for (int j = 0; j < 4; ++j) {
      const int col = bn + wn + j * 16 + frow;
#pragma unroll
      for (int e = 0; e < 8; ++e) {
        float vv = acc[i][j][e];
        if (mode == 1) vv = vv / (1.0f + __expf(-vv));   // SiLU
        C[(size_t)(row0 + e) * N + col] = vv;
      }
    }
  }
}

// ---------------- retention scan (3-pass chunked) ----------------
__device__ __forceinline__ float lam_of(const float* beta, int h) {
  float lam = 1.0f - exp2f(-beta[h]);
  return fminf(fmaxf(lam, 1.1754944e-38f), 1.0f - 1e-9f);
}

// Pass 1: per (b, chunk, h, dh) local scan, in-place over v; record chunk-end state.
__global__ __launch_bounds__(256)
void scan_pass1(float* __restrict__ v, const float* __restrict__ beta,
                float* __restrict__ chunkEnd) {
  int t = blockIdx.x * blockDim.x + threadIdx.x;   // 65536 threads
  int dh = t & 63, h = (t >> 6) & 15, c = (t >> 10) & 15, b = t >> 14;
  float lam = lam_of(beta, h);
  size_t base = (size_t)(b * Ll + c * CL) * Dd + h * DhC + dh;
  float s = 0.0f;
  for (int i = 0; i < CL; ++i) {
    float xv = v[base + (size_t)i * Dd];
    s = lam * s + xv;
    v[base + (size_t)i * Dd] = s;                  // store local prefix state
  }
  chunkEnd[t] = s;                                 // layout [b][c][h][dh]
}

// Pass 2: serial scan over the 16 chunk carries per (b,h,dh).
__global__ __launch_bounds__(256)
void scan_pass2(const float* __restrict__ chunkEnd, const float* __restrict__ beta,
                float* __restrict__ carryIn) {
  int t = blockIdx.x * blockDim.x + threadIdx.x;   // 4096 threads
  int dh = t & 63, h = (t >> 6) & 15, b = t >> 10;
  float lam = lam_of(beta, h);
  float lamP = __powf(lam, (float)CL);
  float run = 0.0f;
  for (int c = 0; c < NC; ++c) {
    size_t idx = ((size_t)(b * NC + c) * Hh + h) * DhC + dh;
    carryIn[idx] = run;
    run = chunkEnd[idx] + lamP * run;
  }
}

// Pass 3: s_t = local_s_t + lam^{i+1} * carry_in;  ret = q * s  (in-place over q)
__global__ __launch_bounds__(256)
void scan_pass3(const float* __restrict__ sLocal, float* __restrict__ q,
                const float* __restrict__ beta, const float* __restrict__ carryIn) {
  int t = blockIdx.x * blockDim.x + threadIdx.x;   // 65536 threads
  int dh = t & 63, h = (t >> 6) & 15, c = (t >> 10) & 15, b = t >> 14;
  float lam = lam_of(beta, h);
  float carry = carryIn[t];
  size_t base = (size_t)(b * Ll + c * CL) * Dd + h * DhC + dh;
  for (int i = 0; i < CL; ++i) {
    carry *= lam;
    size_t idx = base + (size_t)i * Dd;
    q[idx] = q[idx] * (sLocal[idx] + carry);
  }
}

// ---------------- LayerNorm + gate, emits fp16 for final GEMM ----------------
__global__ __launch_bounds__(256)
void ln_gate_kernel(const float* __restrict__ ret, const float* __restrict__ g,
                    const float* __restrict__ gamma, const float* __restrict__ betaLN,
                    half_t* __restrict__ yg) {
  __shared__ float red[256];
  const int row = blockIdx.x;
  const int tid = threadIdx.x;
  const float* r = ret + (size_t)row * Dd;

  float sum = 0.0f;
  for (int j = tid; j < Dd; j += 256) sum += r[j];
  red[tid] = sum; __syncthreads();
  for (int s = 128; s > 0; s >>= 1) { if (tid < s) red[tid] += red[tid + s]; __syncthreads(); }
  float mu = red[0] / (float)Dd;
  __syncthreads();

  float vs = 0.0f;
  for (int j = tid; j < Dd; j += 256) { float d = r[j] - mu; vs += d * d; }
  red[tid] = vs; __syncthreads();
  for (int s = 128; s > 0; s >>= 1) { if (tid < s) red[tid] += red[tid + s]; __syncthreads(); }
  float inv = rsqrtf(red[0] / (float)Dd + 1e-5f);

  for (int j = tid; j < Dd; j += 256) {
    float y = (r[j] - mu) * inv * gamma[j] + betaLN[j];
    yg[(size_t)row * Dd + j] = (half_t)(y * g[(size_t)row * Dd + j]);
  }
}

// ---------------- host-side orchestration ----------------
extern "C" void kernel_launch(void* const* d_in, const int* in_sizes, int n_in,
                              void* d_out, int out_size, void* d_ws, size_t ws_size,
                              hipStream_t stream) {
  const float* x    = (const float*)d_in[0];
  const float* Wq   = (const float*)d_in[1];
  const float* Wv   = (const float*)d_in[2];
  const float* Wo   = (const float*)d_in[3];
  const float* Wg   = (const float*)d_in[4];
  const float* beta = (const float*)d_in[5];
  const float* ln_g = (const float*)d_in[6];
  const float* ln_b = (const float*)d_in[7];
  float* out = (float*)d_out;

  char* ws = (char*)d_ws;
  size_t o = 0;
  half_t* xh  = (half_t*)(ws + o); o += (size_t)Mrows * Dd * sizeof(half_t);  // also reused as yg
  half_t* wqh = (half_t*)(ws + o); o += (size_t)Dd * Dd * sizeof(half_t);
  half_t* wvh = (half_t*)(ws + o); o += (size_t)Dd * Dd * sizeof(half_t);
  half_t* wgh = (half_t*)(ws + o); o += (size_t)Dd * Dd * sizeof(half_t);
  half_t* woh = (half_t*)(ws + o); o += (size_t)Dd * Dd * sizeof(half_t);
  float*  q   = (float*)(ws + o);  o += (size_t)Mrows * Dd * sizeof(float);   // q, then ret (in-place)
  float*  v   = (float*)(ws + o);  o += (size_t)Mrows * Dd * sizeof(float);   // v, then local state (in-place)
  float*  g   = (float*)(ws + o);  o += (size_t)Mrows * Dd * sizeof(float);   // silu(x Wg^T)
  float* cEnd = (float*)(ws + o);  o += (size_t)Bb * NC * Hh * DhC * sizeof(float);
  float* cIn  = (float*)(ws + o);  o += (size_t)Bb * NC * Hh * DhC * sizeof(float);

  const int nX = Mrows * Dd;   // 16777216
  const int nW = Dd * Dd;      // 1048576
  cvt_f32_f16_kernel<<<(nX + 255) / 256, 256, 0, stream>>>(x,  xh,  nX);
  cvt_f32_f16_kernel<<<(nW + 255) / 256, 256, 0, stream>>>(Wq, wqh, nW);
  cvt_f32_f16_kernel<<<(nW + 255) / 256, 256, 0, stream>>>(Wv, wvh, nW);
  cvt_f32_f16_kernel<<<(nW + 255) / 256, 256, 0, stream>>>(Wg, wgh, nW);
  cvt_f32_f16_kernel<<<(nW + 255) / 256, 256, 0, stream>>>(Wo, woh, nW);

  dim3 gg(Mrows / TM, Dd / TN);   // (128, 4)
  gemm_wmma_f16<<<gg, 256, 0, stream>>>(xh, wqh, q, Mrows, Dd, Dd, 0);
  gemm_wmma_f16<<<gg, 256, 0, stream>>>(xh, wvh, v, Mrows, Dd, Dd, 0);
  gemm_wmma_f16<<<gg, 256, 0, stream>>>(xh, wgh, g, Mrows, Dd, Dd, 1);  // fused SiLU

  scan_pass1<<<(Bb * NC * Hh * DhC) / 256, 256, 0, stream>>>(v, beta, cEnd);
  scan_pass2<<<(Bb * Hh * DhC) / 256, 256, 0, stream>>>(cEnd, beta, cIn);
  scan_pass3<<<(Bb * NC * Hh * DhC) / 256, 256, 0, stream>>>(v, q, beta, cIn);

  ln_gate_kernel<<<Mrows, 256, 0, stream>>>(q, g, ln_g, ln_b, xh);       // yg -> xh (fp16)

  gemm_wmma_f16<<<gg, 256, 0, stream>>>(xh, woh, out, Mrows, Dd, Dd, 0);
}
